// SpectralGAT_13666585935938
// MI455X (gfx1250) — matrix operand order repaired and checked by reading, not compiled, not run on hardware
//
#include <hip/hip_runtime.h>
#include <hip/hip_bf16.h>
#include <math.h>

typedef __attribute__((ext_vector_type(16))) __bf16 v16bf;
typedef __attribute__((ext_vector_type(8)))  __bf16 v8bf;
typedef __attribute__((ext_vector_type(8)))  float  v8f;

#define NEG_SLOPE 0.2f

// ---------------- wave helpers (wave32) ----------------
__device__ __forceinline__ float warp_sum(float v) {
#pragma unroll
  for (int m = 16; m > 0; m >>= 1) v += __shfl_xor(v, m, 32);
  return v;
}

// ordered-uint encoding for float atomicMax (handles negatives)
__device__ __forceinline__ unsigned enc_f(float f) {
  unsigned u = __float_as_uint(f);
  return (u & 0x80000000u) ? ~u : (u | 0x80000000u);
}
__device__ __forceinline__ float dec_f(unsigned u) {
  return (u & 0x80000000u) ? __uint_as_float(u & 0x7fffffffu)
                           : __uint_as_float(~u);
}

// ---------------- cast f32 -> bf16 ----------------
__global__ void cast_bf16_kernel(const float* __restrict__ in,
                                 __bf16* __restrict__ out, long n) {
  long i = (long)blockIdx.x * blockDim.x + threadIdx.x;
  if (i < n) out[i] = (__bf16)in[i];
}

// ---------------- pack B (row-major KxN, bf16) into WMMA fragment order ----------
// Bp[((tk*tilesN16 + tn)*32 + lane)*16 + i] = B[(tk*32 + (lane>>4)*16 + i)*N + tn*16 + (lane&15)]
// One thread per (tile, lane); each writes 16 contiguous bf16 (32 bytes).
__global__ void pack_b_kernel(const __bf16* __restrict__ B, __bf16* __restrict__ Bp,
                              int K, int Ncols) {
  int tilesN = Ncols >> 4;
  int tilesK = K >> 5;
  int t = blockIdx.x * blockDim.x + threadIdx.x;
  int tot = tilesK * tilesN * 32;
  if (t >= tot) return;
  int lane = t & 31;
  int tile = t >> 5;
  int tk = tile / tilesN, tn = tile - tk * tilesN;
  int n  = tn * 16 + (lane & 15);
  int k0 = tk * 32 + (lane >> 4) * 16;
  __bf16* dst = Bp + (size_t)t * 16;
#pragma unroll
  for (int i = 0; i < 16; ++i) dst[i] = B[(size_t)(k0 + i) * Ncols + n];
}

// ---------------- WMMA fragment loads (wide b128) ----------------
// A: row-major MxK bf16. Lane's 16 values = one contiguous 16-element run (K = 16*(lane>>4) ..).
__device__ __forceinline__ v16bf load_a_frag(const __bf16* __restrict__ A, int lda,
                                             int row0, int k0, int lane) {
  int m  = row0 + (lane & 15);
  int kb = (lane >> 4) * 16;
  const v8bf* p = (const v8bf*)(A + (size_t)m * lda + k0 + kb);
  v8bf lo = p[0];   // K = kb .. kb+7
  v8bf hi = p[1];   // K = kb+8 .. kb+15
  return __builtin_shufflevector(lo, hi, 0, 1, 2, 3, 4, 5, 6, 7,
                                 8, 9, 10, 11, 12, 13, 14, 15);
}

// B: packed fragment order; lane reads its own contiguous 32 bytes.
__device__ __forceinline__ v16bf load_b_packed(const __bf16* __restrict__ Bp,
                                               int tilesN16, int tk, int tn16, int lane) {
  const v8bf* p = (const v8bf*)(Bp + ((((size_t)tk * tilesN16 + tn16) * 32 + lane) << 4));
  v8bf lo = p[0];
  v8bf hi = p[1];
  return __builtin_shufflevector(lo, hi, 0, 1, 2, 3, 4, 5, 6, 7,
                                 8, 9, 10, 11, 12, 13, 14, 15);
}

// ---------------- WMMA GEMM: C(f32, MxN) = A(bf16 MxK) * Bp(packed) ----------------
// One wave computes a 16x64 tile (4 accumulators reuse one A fragment).
__global__ void gemm_bf16_wmma_kernel(const __bf16* __restrict__ A,
                                      const __bf16* __restrict__ Bp,
                                      float* __restrict__ C,
                                      int M, int K, int Ncols) {
  int wave = (int)((blockIdx.x * blockDim.x + threadIdx.x) >> 5);
  int lane = threadIdx.x & 31;
  int tilesW = Ncols >> 6;      // 64-col wave tiles
  int tilesN16 = Ncols >> 4;    // 16-col pack tiles
  int tm = wave / tilesW;
  int tn = wave - tm * tilesW;
  if (tm * 16 >= M) return;
  int row0 = tm * 16;
  int tn16 = tn * 4;

  v8f c0 = {}, c1 = {}, c2 = {}, c3 = {};
#pragma unroll 2
  for (int k0 = 0; k0 < K; k0 += 32) {
    int tk = k0 >> 5;
    v16bf a  = load_a_frag(A, K, row0, k0, lane);
    v16bf b0 = load_b_packed(Bp, tilesN16, tk, tn16 + 0, lane);
    v16bf b1 = load_b_packed(Bp, tilesN16, tk, tn16 + 1, lane);
    v16bf b2 = load_b_packed(Bp, tilesN16, tk, tn16 + 2, lane);
    v16bf b3 = load_b_packed(Bp, tilesN16, tk, tn16 + 3, lane);
    c0 = __builtin_amdgcn_wmma_f32_16x16x32_bf16(false, a, false, b0, (short)0, c0, false, false);
    c1 = __builtin_amdgcn_wmma_f32_16x16x32_bf16(false, a, false, b1, (short)0, c1, false, false);
    c2 = __builtin_amdgcn_wmma_f32_16x16x32_bf16(false, a, false, b2, (short)0, c2, false, false);
    c3 = __builtin_amdgcn_wmma_f32_16x16x32_bf16(false, a, false, b3, (short)0, c3, false, false);
  }
  // C/D 16x16 f32 layout: col = lane&15, VGPR r -> row (lane>>4)*8 + r
  int n  = lane & 15;
  int mb = (lane >> 4) * 8;
#pragma unroll
  for (int r = 0; r < 8; ++r) {
    float* crow = C + (size_t)(row0 + mb + r) * Ncols + tn * 64;
    crow[n]      = c0[r];
    crow[16 + n] = c1[r];
    crow[32 + n] = c2[r];
    crow[48 + n] = c3[r];
  }
}

// ---------------- per-node attention scores: s = <h[n,h,:], a[h,:]> ----------------
__global__ void attn_scores_kernel(const float* __restrict__ h,
                                   const float* __restrict__ a_src,
                                   const float* __restrict__ a_dst,
                                   float* __restrict__ ssrc, float* __restrict__ sdst,
                                   int N, int H, int C) {
  int wv   = (int)((blockIdx.x * blockDim.x + threadIdx.x) >> 5);
  int lane = threadIdx.x & 31;
  if (wv >= N) return;
  const float* row = h + (size_t)wv * H * C;
  for (int hh = 0; hh < H; ++hh) {
    float ss = 0.f, sd = 0.f;
    for (int c = lane; c < C; c += 32) {
      float v = row[hh * C + c];
      ss += v * a_src[hh * C + c];
      sd += v * a_dst[hh * C + c];
    }
    ss = warp_sum(ss);
    sd = warp_sum(sd);
    if (lane == 0) { ssrc[wv * H + hh] = ss; sdst[wv * H + hh] = sd; }
  }
}

// ---------------- edge pass A: segment max ----------------
__global__ void edge_max_kernel(const int* __restrict__ src, const int* __restrict__ dst,
                                const float* __restrict__ ssrc, const float* __restrict__ sdst,
                                unsigned* __restrict__ emax, int E, int N, int H) {
  int tid = blockIdx.x * blockDim.x + threadIdx.x;
  int tot = (E + N) * H;
  if (tid >= tot) return;
  int e = tid / H, hh = tid - e * H;
  int s = (e < E) ? src[e] : (e - E);
  int d = (e < E) ? dst[e] : (e - E);
  float x = ssrc[s * H + hh] + sdst[d * H + hh];
  x = (x > 0.f) ? x : NEG_SLOPE * x;
  atomicMax(&emax[d * H + hh], enc_f(x));
}

// ---------------- edge pass B: exp + segment sum ----------------
__global__ void edge_sum_kernel(const int* __restrict__ src, const int* __restrict__ dst,
                                const float* __restrict__ ssrc, const float* __restrict__ sdst,
                                const unsigned* __restrict__ emax,
                                float* __restrict__ we, float* __restrict__ denom,
                                int E, int N, int H) {
  int tid = blockIdx.x * blockDim.x + threadIdx.x;
  int tot = (E + N) * H;
  if (tid >= tot) return;
  int e = tid / H, hh = tid - e * H;
  int s = (e < E) ? src[e] : (e - E);
  int d = (e < E) ? dst[e] : (e - E);
  float x = ssrc[s * H + hh] + sdst[d * H + hh];
  x = (x > 0.f) ? x : NEG_SLOPE * x;
  float w = __expf(x - dec_f(emax[d * H + hh]));
  we[tid] = w;
  atomicAdd(&denom[d * H + hh], w);
}

// ---------------- edge pass C: weighted scatter-aggregate (wave per edge) ----------------
__global__ void edge_agg_kernel(const int* __restrict__ src, const int* __restrict__ dst,
                                const float* __restrict__ we, const float* __restrict__ denom,
                                const float* __restrict__ feat, float* __restrict__ agg,
                                int E, int N, int H, int C) {
  int wv   = (int)((blockIdx.x * blockDim.x + threadIdx.x) >> 5);
  int lane = threadIdx.x & 31;
  int tot = E + N;
  if (wv >= tot) return;
  int s = (wv < E) ? src[wv] : (wv - E);
  int d = (wv < E) ? dst[wv] : (wv - E);
  for (int hh = 0; hh < H; ++hh) {
    float alpha = we[(size_t)wv * H + hh] / (denom[d * H + hh] + 1e-16f);
    const float* fsrc = feat + ((size_t)s * H + hh) * C;
    float* adst = agg + ((size_t)d * H + hh) * C;
    for (int c = lane; c < C; c += 32)
      atomicAdd(&adst[c], alpha * fsrc[c]);
  }
}

// ---------------- bias + ELU, optional f32 and bf16 outputs ----------------
__global__ void bias_elu_kernel(const float* __restrict__ agg, const float* __restrict__ bias,
                                float* __restrict__ out_f, __bf16* __restrict__ out_b,
                                long n, int HC) {
  long i = (long)blockIdx.x * blockDim.x + threadIdx.x;
  if (i >= n) return;
  float v = agg[i] + bias[(int)(i % HC)];
  v = (v > 0.f) ? v : (__expf(v) - 1.f);
  if (out_f) out_f[i] = v;
  if (out_b) out_b[i] = (__bf16)v;
}

// ---------------- MLP: [64->32] + LayerNorm + ReLU + [32->16], wave per node ----------------
__global__ void mlp_kernel(const float* __restrict__ h3,
                           const float* __restrict__ Wm1, const float* __restrict__ bm1,
                           const float* __restrict__ g_ln, const float* __restrict__ b_ln,
                           const float* __restrict__ Wm2, const float* __restrict__ bm2,
                           float* __restrict__ out, int N) {
  __shared__ float rowbuf[4][64];
  __shared__ float zbuf[4][32];
  int w    = threadIdx.x >> 5;
  int lane = threadIdx.x & 31;
  int node = blockIdx.x * 4 + w;
  bool valid = node < N;
  if (valid) {
    rowbuf[w][lane]      = h3[(size_t)node * 64 + lane];
    rowbuf[w][lane + 32] = h3[(size_t)node * 64 + lane + 32];
  }
  __syncthreads();
  float y = 0.f;
  if (valid) {
#pragma unroll 8
    for (int k = 0; k < 64; ++k) y += rowbuf[w][k] * Wm1[k * 32 + lane];
    y += bm1[lane];
  }
  float mu  = warp_sum(y) * (1.f / 32.f);
  float dv  = y - mu;
  float var = warp_sum(dv * dv) * (1.f / 32.f);
  float z = dv * rsqrtf(var + 1e-5f) * g_ln[lane] + b_ln[lane];
  z = fmaxf(z, 0.f);
  zbuf[w][lane] = z;
  __syncthreads();
  if (valid && lane < 16) {
    float o = bm2[lane];
#pragma unroll 8
    for (int j = 0; j < 32; ++j) o += zbuf[w][j] * Wm2[j * 16 + lane];
    out[(size_t)node * 16 + lane] = o;
  }
}

// =====================================================================
extern "C" void kernel_launch(void* const* d_in, const int* in_sizes, int n_in,
                              void* d_out, int out_size, void* d_ws, size_t ws_size,
                              hipStream_t stream) {
  const float* x      = (const float*)d_in[0];
  const int*   ei     = (const int*)d_in[1];
  const float* W1     = (const float*)d_in[3];
  const float* a_src1 = (const float*)d_in[4];
  const float* a_dst1 = (const float*)d_in[5];
  const float* b1     = (const float*)d_in[6];
  const float* W2     = (const float*)d_in[7];
  const float* a_src2 = (const float*)d_in[8];
  const float* a_dst2 = (const float*)d_in[9];
  const float* b2     = (const float*)d_in[10];
  const float* Wm1    = (const float*)d_in[11];
  const float* bm1    = (const float*)d_in[12];
  const float* g_ln   = (const float*)d_in[13];
  const float* b_ln   = (const float*)d_in[14];
  const float* Wm2    = (const float*)d_in[15];
  const float* bm2    = (const float*)d_in[16];
  float* out = (float*)d_out;

  const int F_in = 128, H1 = 4, C1 = 128, HC1 = 512, C2 = 64;
  const int N = in_sizes[0] / F_in;
  const int E = in_sizes[1] / 2;
  const int* srcIdx = ei;
  const int* dstIdx = ei + E;

  // ---- bump allocator over d_ws ----
  char* wp = (char*)d_ws;
  auto alloc = [&](size_t bytes) -> void* {
    void* p = (void*)wp;
    wp += (bytes + 255) & ~(size_t)255;
    return p;
  };
  __bf16*  xb     = (__bf16*)alloc((size_t)N * F_in * 2);
  __bf16*  W1b    = (__bf16*)alloc((size_t)F_in * HC1 * 2);
  __bf16*  W1p    = (__bf16*)alloc((size_t)F_in * HC1 * 2);
  float*   h1     = (float*) alloc((size_t)N * HC1 * 4);
  float*   ssrc1  = (float*) alloc((size_t)N * H1 * 4);
  float*   sdst1  = (float*) alloc((size_t)N * H1 * 4);
  unsigned* emax1 = (unsigned*)alloc((size_t)N * H1 * 4);
  float*   den1   = (float*) alloc((size_t)N * H1 * 4);
  float*   we1    = (float*) alloc((size_t)(E + N) * H1 * 4);
  float*   agg1   = (float*) alloc((size_t)N * HC1 * 4);
  __bf16*  hb     = (__bf16*)alloc((size_t)N * HC1 * 2);
  __bf16*  W2b    = (__bf16*)alloc((size_t)HC1 * C2 * 2);
  __bf16*  W2p    = (__bf16*)alloc((size_t)HC1 * C2 * 2);
  float*   h2     = (float*) alloc((size_t)N * C2 * 4);
  float*   ssrc2  = (float*) alloc((size_t)N * 4);
  float*   sdst2  = (float*) alloc((size_t)N * 4);
  unsigned* emax2 = (unsigned*)alloc((size_t)N * 4);
  float*   den2   = (float*) alloc((size_t)N * 4);
  float*   we2    = (float*) alloc((size_t)(E + N) * 4);
  float*   agg2   = (float*) alloc((size_t)N * C2 * 4);
  float*   h3     = (float*) alloc((size_t)N * C2 * 4);

  auto blk = [](long n, int t) { return (unsigned)((n + t - 1) / t); };

  // zero accumulators / maxima (graph-capture safe)
  hipMemsetAsync(emax1, 0, (size_t)N * H1 * 4, stream);
  hipMemsetAsync(den1,  0, (size_t)N * H1 * 4, stream);
  hipMemsetAsync(agg1,  0, (size_t)N * HC1 * 4, stream);
  hipMemsetAsync(emax2, 0, (size_t)N * 4, stream);
  hipMemsetAsync(den2,  0, (size_t)N * 4, stream);
  hipMemsetAsync(agg2,  0, (size_t)N * C2 * 4, stream);

  // ---- conv1 ----
  cast_bf16_kernel<<<blk((long)N * F_in, 256), 256, 0, stream>>>(x, xb, (long)N * F_in);
  cast_bf16_kernel<<<blk((long)F_in * HC1, 256), 256, 0, stream>>>(W1, W1b, (long)F_in * HC1);
  pack_b_kernel<<<blk((long)(F_in / 32) * (HC1 / 16) * 32, 256), 256, 0, stream>>>(W1b, W1p, F_in, HC1);
  {
    long waves = (long)((N + 15) / 16) * (HC1 / 64);
    gemm_bf16_wmma_kernel<<<blk(waves * 32, 256), 256, 0, stream>>>(xb, W1p, h1, N, F_in, HC1);
  }
  attn_scores_kernel<<<blk((long)N * 32, 256), 256, 0, stream>>>(h1, a_src1, a_dst1, ssrc1, sdst1, N, H1, C1);
  edge_max_kernel<<<blk((long)(E + N) * H1, 256), 256, 0, stream>>>(srcIdx, dstIdx, ssrc1, sdst1, emax1, E, N, H1);
  edge_sum_kernel<<<blk((long)(E + N) * H1, 256), 256, 0, stream>>>(srcIdx, dstIdx, ssrc1, sdst1, emax1, we1, den1, E, N, H1);
  edge_agg_kernel<<<blk((long)(E + N) * 32, 256), 256, 0, stream>>>(srcIdx, dstIdx, we1, den1, h1, agg1, E, N, H1, C1);
  bias_elu_kernel<<<blk((long)N * HC1, 256), 256, 0, stream>>>(agg1, b1, (float*)nullptr, hb, (long)N * HC1, HC1);

  // ---- conv2 (H=1, mean over heads is identity) ----
  cast_bf16_kernel<<<blk((long)HC1 * C2, 256), 256, 0, stream>>>(W2, W2b, (long)HC1 * C2);
  pack_b_kernel<<<blk((long)(HC1 / 32) * (C2 / 16) * 32, 256), 256, 0, stream>>>(W2b, W2p, HC1, C2);
  {
    long waves = (long)((N + 15) / 16) * (C2 / 64);
    gemm_bf16_wmma_kernel<<<blk(waves * 32, 256), 256, 0, stream>>>(hb, W2p, h2, N, HC1, C2);
  }
  attn_scores_kernel<<<blk((long)N * 32, 256), 256, 0, stream>>>(h2, a_src2, a_dst2, ssrc2, sdst2, N, 1, C2);
  edge_max_kernel<<<blk((long)(E + N), 256), 256, 0, stream>>>(srcIdx, dstIdx, ssrc2, sdst2, emax2, E, N, 1);
  edge_sum_kernel<<<blk((long)(E + N), 256), 256, 0, stream>>>(srcIdx, dstIdx, ssrc2, sdst2, emax2, we2, den2, E, N, 1);
  edge_agg_kernel<<<blk((long)(E + N) * 32, 256), 256, 0, stream>>>(srcIdx, dstIdx, we2, den2, h2, agg2, E, N, 1, C2);
  bias_elu_kernel<<<blk((long)N * C2, 256), 256, 0, stream>>>(agg2, b2, h3, (__bf16*)nullptr, (long)N * C2, C2);

  // ---- MLP head ----
  mlp_kernel<<<blk((long)N * 32, 128), 128, 0, stream>>>(h3, Wm1, bm1, g_ln, b_ln, Wm2, bm2, out, N);
}
